// EncodeProcessDecode_58085137711594
// MI455X (gfx1250) — matrix operand order repaired and checked by reading, not compiled, not run on hardware
//
#include <hip/hip_runtime.h>
#include <hip/hip_bf16.h>
#include <cstdint>
#include <cstddef>

typedef __attribute__((ext_vector_type(16))) _Float16 v16h;
typedef __attribute__((ext_vector_type(8)))  float    v8f;

namespace {
constexpr int N_NODES = 50000;
constexpr int N_EDGES = 300000;
constexpr int H       = 128;
constexpr int P_STEPS = 3;
constexpr int KMAX    = 384;   // max padded layer-1 K (3*H for the edge MLP)
}

// ------------------------------------------------------------------ zero fill
__global__ void __launch_bounds__(256) k_zero(float* p, int n) {
  int i = blockIdx.x * 256 + threadIdx.x;
  if (i < n) p[i] = 0.f;
}

// ---------------- pack f32 weight [Krows x Ncols] into WMMA B fragments (f16)
// kt-major layout: out[((kt*Nt + nb)*32 + lane)*16 + i] = W[K][Ncol] with
//   K = kt*32 + (lane<16?0:8) + (i<8 ? i : 8+i) ,  Ncol = nb*16 + (lane&15) ; OOB -> 0
__global__ void __launch_bounds__(256)
k_pack_w(const float* __restrict__ W, int Krows, int Ncols, int Kt, int Nt,
         _Float16* __restrict__ out) {
  int t = blockIdx.x * 256 + threadIdx.x;
  int total = Nt * Kt * 512;
  if (t >= total) return;
  int i    = t & 15;
  int lane = (t >> 4) & 31;
  int nb   = (t >> 9) % Nt;
  int kt   = (t >> 9) / Nt;
  int K = kt * 32 + ((lane < 16) ? 0 : 8) + ((i < 8) ? i : 8 + i);
  int n = nb * 16 + (lane & 15);
  float v = (K < Krows && n < Ncols) ? W[(size_t)K * Ncols + n] : 0.f;
  out[t] = (_Float16)v;
}

// --------------------------------------------------------------- fused MLP
// Per wave: 16 rows. A = concat of up to 3 gathered f32 segments, staged as
// f16 in LDS. Y = [LN]( relu(A@W1+b1) @ W2 + b2 ) [+ resid]; direct store or
// scatter atomicAdd (segment_sum). 2 waves / 64 threads per block.
// K-outer / N-inner schedule: each A fragment is loaded from LDS once and fed
// to 8 independent WMMAs against the streaming (L2-resident) weight fragments.
__global__ void __launch_bounds__(64)
k_fused_mlp(int nRows,
            const float* __restrict__ sA, const int* __restrict__ iA, int wA,
            const float* __restrict__ sB, const int* __restrict__ iB, int wB,
            const float* __restrict__ sC, const int* __restrict__ iC, int wC,
            int Kt1,
            const _Float16* __restrict__ W1p, const float* __restrict__ bias1,
            const _Float16* __restrict__ W2p, const float* __restrict__ bias2,
            int Nt2, int outCols,
            const float* __restrict__ lnG, const float* __restrict__ lnB,
            const float* __restrict__ resid,
            const int* __restrict__ sIdx,
            float* __restrict__ outp, int outStride)
{
  __shared__ __align__(16) _Float16 Ah[2][16][KMAX];  // 24 KB (reused as f32 out tile)
  __shared__ __align__(16) _Float16 Hh[2][16][H];     // 8 KB hidden (f16)
  __shared__ float Mu[2][16];
  __shared__ float Rs[2][16];

  const int wave     = threadIdx.x >> 5;
  const int lane     = threadIdx.x & 31;
  const int tileBase = blockIdx.x * 32 + wave * 16;
  const int K1       = Kt1 * 32;
  const int hiOff    = (lane >> 4) << 3;   // 0 for lanes 0..15, 8 for 16..31
  const int nlo      = lane & 15;

  // ---- stage A rows: gather + f32->f16, zero pad ----
  {
    int r = lane / K1, k = lane - r * K1;   // K1 >= 32 so per-step carry is <= 1
    for (int t = lane; t < 16 * K1; t += 32) {
      float v = 0.f;
      int gr = tileBase + r;
      if (gr < nRows) {
        if (k < wA) {
          int rr = iA ? iA[gr] : gr;
          v = sA[(size_t)rr * wA + k];
        } else if (k < wA + wB) {
          int rr = iB ? iB[gr] : gr;
          v = sB[(size_t)rr * wB + (k - wA)];
        } else if (k < wA + wB + wC) {
          int rr = iC ? iC[gr] : gr;
          v = sC[(size_t)rr * wC + (k - wA - wB)];
        }
      }
      Ah[wave][r][k] = (_Float16)v;
      k += 32;
      if (k >= K1) { k -= K1; r += 1; }
    }
  }

  // ---- layer 1: hidden = relu(A @ W1 + b1) -> LDS f16 ----
  {
    v8f acc[8];
    #pragma unroll
    for (int nb = 0; nb < 8; ++nb)
      acc[nb] = (v8f){0.f, 0.f, 0.f, 0.f, 0.f, 0.f, 0.f, 0.f};
    for (int kt = 0; kt < Kt1; ++kt) {
      v16h a;
      const _Float16* ap = &Ah[wave][nlo][kt * 32 + hiOff];
      #pragma unroll
      for (int i = 0; i < 8; ++i) { a[i] = ap[i]; a[i + 8] = ap[i + 16]; }
      const _Float16* wp = &W1p[((size_t)kt * 8) * 512 + (size_t)lane * 16];
      __builtin_prefetch(wp + 8 * 512, 0, 1);   // next K-tile of weight stream
      #pragma unroll
      for (int nb = 0; nb < 8; ++nb) {
        v16h b = *(const v16h*)(wp + (size_t)nb * 512);
        acc[nb] = __builtin_amdgcn_wmma_f32_16x16x32_f16(false, a, false, b,
                                                         (short)0, acc[nb],
                                                         false, false);
      }
    }
    #pragma unroll
    for (int nb = 0; nb < 8; ++nb) {
      int nc = nb * 16 + nlo;
      float bb = bias1[nc];
      #pragma unroll
      for (int j = 0; j < 8; ++j) {
        float h = acc[nb][j] + bb;
        Hh[wave][hiOff + j][nc] = (_Float16)(h > 0.f ? h : 0.f);
      }
    }
  }

  // ---- layer 2: Ot = hidden @ W2 + b2 (f32 tile in LDS, aliased on Ah) ----
  float* Ot = (float*)&Ah[wave][0][0];     // 16 x (Nt2*16) f32, fits in 12 KB
  const int ncol2 = Nt2 * 16;
  if (Nt2 == 8) {
    v8f acc[8];
    #pragma unroll
    for (int nb = 0; nb < 8; ++nb)
      acc[nb] = (v8f){0.f, 0.f, 0.f, 0.f, 0.f, 0.f, 0.f, 0.f};
    #pragma unroll
    for (int kt = 0; kt < 4; ++kt) {       // K = H = 128
      v16h a;
      const _Float16* ap = &Hh[wave][nlo][kt * 32 + hiOff];
      #pragma unroll
      for (int i = 0; i < 8; ++i) { a[i] = ap[i]; a[i + 8] = ap[i + 16]; }
      const _Float16* wp = &W2p[((size_t)kt * 8) * 512 + (size_t)lane * 16];
      #pragma unroll
      for (int nb = 0; nb < 8; ++nb) {
        v16h b = *(const v16h*)(wp + (size_t)nb * 512);
        acc[nb] = __builtin_amdgcn_wmma_f32_16x16x32_f16(false, a, false, b,
                                                         (short)0, acc[nb],
                                                         false, false);
      }
    }
    #pragma unroll
    for (int nb = 0; nb < 8; ++nb) {
      int nc = nb * 16 + nlo;
      float bb = bias2[nc];
      #pragma unroll
      for (int j = 0; j < 8; ++j)
        Ot[(hiOff + j) * 128 + nc] = acc[nb][j] + bb;
    }
  } else {                                  // Nt2 == 1 (decoder head)
    v8f acc = (v8f){0.f, 0.f, 0.f, 0.f, 0.f, 0.f, 0.f, 0.f};
    #pragma unroll
    for (int kt = 0; kt < 4; ++kt) {
      v16h a;
      const _Float16* ap = &Hh[wave][nlo][kt * 32 + hiOff];
      #pragma unroll
      for (int i = 0; i < 8; ++i) { a[i] = ap[i]; a[i + 8] = ap[i + 16]; }
      v16h b = *(const v16h*)&W2p[((size_t)kt * 32 + lane) * 16];
      acc = __builtin_amdgcn_wmma_f32_16x16x32_f16(false, a, false, b,
                                                   (short)0, acc, false, false);
    }
    int nc = nlo;
    float bb = (nc < outCols) ? bias2[nc] : 0.f;
    #pragma unroll
    for (int j = 0; j < 8; ++j)
      Ot[(hiOff + j) * 16 + nc] = acc[j] + bb;
  }

  // ---- epilogue: LayerNorm / residual / store or scatter ----
  if (lnG) {
    if (lane < 16) {
      float s = 0.f, ss = 0.f;
      for (int c = 0; c < ncol2; ++c) {
        float v = Ot[lane * ncol2 + c];
        s += v; ss += v * v;
      }
      float m   = s / (float)ncol2;
      float var = ss / (float)ncol2 - m * m;
      Mu[wave][lane] = m;
      Rs[wave][lane] = rsqrtf(var + 1e-5f);
    }
    // wave-synchronous LDS: per-wave DS ops complete in order
    for (int t = lane; t < 16 * ncol2; t += 32) {
      int r = t / ncol2, c = t - r * ncol2;
      int gr = tileBase + r;
      if (gr >= nRows) continue;
      float v = (Ot[t] - Mu[wave][r]) * Rs[wave][r] * lnG[c] + lnB[c];
      if (resid) v += resid[(size_t)gr * H + c];
      if (sIdx)  atomicAdd(&outp[(size_t)sIdx[gr] * H + c], v);
      else       outp[(size_t)gr * outStride + c] = v;
    }
  } else {
    for (int t = lane; t < 16 * ncol2; t += 32) {
      int r = t / ncol2, c = t - r * ncol2;
      int gr = tileBase + r;
      if (gr < nRows && c < outCols)
        outp[(size_t)gr * outStride + c] = Ot[t];
    }
  }
}

// ============================================================== host launcher
extern "C" void kernel_launch(void* const* d_in, const int* in_sizes, int n_in,
                              void* d_out, int out_size, void* d_ws, size_t ws_size,
                              hipStream_t stream) {
  (void)in_sizes; (void)n_in; (void)out_size; (void)ws_size;
  const int N = N_NODES, E = N_EDGES, P = P_STEPS;

  const float* x        = (const float*)d_in[0];
  const float* e        = (const float*)d_in[1];
  const int*   ei       = (const int*)  d_in[2];
  const int*   row      = ei;        // source (x_j)
  const int*   col      = ei + E;    // target (x_i), scatter index
  const float* enc_n_w1 = (const float*)d_in[3];
  const float* enc_n_b1 = (const float*)d_in[4];
  const float* enc_n_w2 = (const float*)d_in[5];
  const float* enc_n_b2 = (const float*)d_in[6];
  const float* enc_n_g  = (const float*)d_in[7];
  const float* enc_n_be = (const float*)d_in[8];
  const float* enc_e_w1 = (const float*)d_in[9];
  const float* enc_e_b1 = (const float*)d_in[10];
  const float* enc_e_w2 = (const float*)d_in[11];
  const float* enc_e_b2 = (const float*)d_in[12];
  const float* enc_e_g  = (const float*)d_in[13];
  const float* enc_e_be = (const float*)d_in[14];
  const float* pe_w1    = (const float*)d_in[15];
  const float* pe_b1    = (const float*)d_in[16];
  const float* pe_w2    = (const float*)d_in[17];
  const float* pe_b2    = (const float*)d_in[18];
  const float* pe_g     = (const float*)d_in[19];
  const float* pe_be    = (const float*)d_in[20];
  const float* pn_w1    = (const float*)d_in[21];
  const float* pn_b1    = (const float*)d_in[22];
  const float* pn_w2    = (const float*)d_in[23];
  const float* pn_b2    = (const float*)d_in[24];
  const float* pn_g     = (const float*)d_in[25];
  const float* pn_be    = (const float*)d_in[26];
  const float* dec_w1   = (const float*)d_in[27];
  const float* dec_b1   = (const float*)d_in[28];
  const float* dec_w2   = (const float*)d_in[29];
  const float* dec_b2   = (const float*)d_in[30];

  // -------- workspace carve --------
  char* base = (char*)d_ws;
  size_t off = 0;
  auto carve = [&](size_t bytes) -> void* {
    void* p = base + off;
    off += (bytes + 255) & ~(size_t)255;
    return p;
  };
  float* xh   = (float*)carve((size_t)N * H * 4);   // node latent (f32 master)
  float* eh   = (float*)carve((size_t)E * H * 4);   // edge latent (f32 master)
  float* aggr = (float*)carve((size_t)N * H * 4);   // segment_sum target
  _Float16* p_enw1 = (_Float16*)carve((size_t)8 * 1  * 512 * 2);
  _Float16* p_enw2 = (_Float16*)carve((size_t)8 * 4  * 512 * 2);
  _Float16* p_eew1 = (_Float16*)carve((size_t)8 * 1  * 512 * 2);
  _Float16* p_eew2 = (_Float16*)carve((size_t)8 * 4  * 512 * 2);
  _Float16* p_pew1 = (_Float16*)carve((size_t)P * 8 * 12 * 512 * 2);
  _Float16* p_pew2 = (_Float16*)carve((size_t)P * 8 * 4  * 512 * 2);
  _Float16* p_pnw1 = (_Float16*)carve((size_t)P * 8 * 8  * 512 * 2);
  _Float16* p_pnw2 = (_Float16*)carve((size_t)P * 8 * 4  * 512 * 2);
  _Float16* p_dw1  = (_Float16*)carve((size_t)8 * 4 * 512 * 2);
  _Float16* p_dw2  = (_Float16*)carve((size_t)1 * 4 * 512 * 2);

  auto pack = [&](const float* W, int K, int Nc, int Kt, int Nt, _Float16* outp) {
    int tot = Nt * Kt * 512;
    k_pack_w<<<dim3((tot + 255) / 256), dim3(256), 0, stream>>>(W, K, Nc, Kt, Nt, outp);
  };
  const int szW1e = 8 * 12 * 512, szW2 = 8 * 4 * 512, szW1n = 8 * 8 * 512;

  pack(enc_n_w1, 15, H, 1, 8, p_enw1);
  pack(enc_n_w2, H,  H, 4, 8, p_enw2);
  pack(enc_e_w1, 7,  H, 1, 8, p_eew1);
  pack(enc_e_w2, H,  H, 4, 8, p_eew2);
  for (int p = 0; p < P; ++p) {
    pack(pe_w1 + (size_t)p * 3 * H * H, 3 * H, H, 12, 8, p_pew1 + (size_t)p * szW1e);
    pack(pe_w2 + (size_t)p * H * H,     H,     H, 4,  8, p_pew2 + (size_t)p * szW2);
    pack(pn_w1 + (size_t)p * 2 * H * H, 2 * H, H, 8,  8, p_pnw1 + (size_t)p * szW1n);
    pack(pn_w2 + (size_t)p * H * H,     H,     H, 4,  8, p_pnw2 + (size_t)p * szW2);
  }
  pack(dec_w1, H, H, 4, 8, p_dw1);
  pack(dec_w2, H, 3, 4, 1, p_dw2);

  const dim3 blk(64);
  const dim3 gN((N + 31) / 32), gE((E + 31) / 32);

  // -------- encoders --------
  k_fused_mlp<<<gN, blk, 0, stream>>>(N,
      x, nullptr, 15,  nullptr, nullptr, 0,  nullptr, nullptr, 0,
      1, p_enw1, enc_n_b1, p_enw2, enc_n_b2,
      8, H, enc_n_g, enc_n_be, nullptr, nullptr, xh, H);
  k_fused_mlp<<<gE, blk, 0, stream>>>(E,
      e, nullptr, 7,   nullptr, nullptr, 0,  nullptr, nullptr, 0,
      1, p_eew1, enc_e_b1, p_eew2, enc_e_b2,
      8, H, enc_e_g, enc_e_be, nullptr, nullptr, eh, H);

  // -------- P message-passing steps --------
  for (int p = 0; p < P; ++p) {
    const _Float16* w1e = p_pew1 + (size_t)p * szW1e;
    const _Float16* w2e = p_pew2 + (size_t)p * szW2;
    const _Float16* w1n = p_pnw1 + (size_t)p * szW1n;
    const _Float16* w2n = p_pnw2 + (size_t)p * szW2;

    k_zero<<<dim3((N * H + 255) / 256), dim3(256), 0, stream>>>(aggr, N * H);

    // message: LN(mlp(cat[xh[col], xh[row], eh])) -> scatter-add onto col
    k_fused_mlp<<<gE, blk, 0, stream>>>(E,
        xh, col, H,  xh, row, H,  eh, nullptr, H,
        12, w1e, pe_b1 + p * H, w2e, pe_b2 + p * H,
        8, H, pe_g + p * H, pe_be + p * H, nullptr, col, aggr, H);

    // edge update: LN(mlp(cat[xh[row], xh[col], eh])) + eh -> eh (in place)
    k_fused_mlp<<<gE, blk, 0, stream>>>(E,
        xh, row, H,  xh, col, H,  eh, nullptr, H,
        12, w1e, pe_b1 + p * H, w2e, pe_b2 + p * H,
        8, H, pe_g + p * H, pe_be + p * H, eh, nullptr, eh, H);

    // node update: LN(mlp(cat[aggr, xh])) + xh -> xh (in place)
    k_fused_mlp<<<gN, blk, 0, stream>>>(N,
        aggr, nullptr, H,  xh, nullptr, H,  nullptr, nullptr, 0,
        8, w1n, pn_b1 + p * H, w2n, pn_b2 + p * H,
        8, H, pn_g + p * H, pn_be + p * H, xh, nullptr, xh, H);
  }

  // -------- decoder (no LN), out = [N,3] f32 --------
  k_fused_mlp<<<gN, blk, 0, stream>>>(N,
      xh, nullptr, H,  nullptr, nullptr, 0,  nullptr, nullptr, 0,
      4, p_dw1, dec_b1, p_dw2, dec_b2,
      1, 3, nullptr, nullptr, nullptr, nullptr, (float*)d_out, 3);
}
